// RelationNetwork_67061619360306
// MI455X (gfx1250) — matrix operand order repaired and checked by reading, not compiled, not run on hardware
//
#include <hip/hip_runtime.h>

typedef __attribute__((ext_vector_type(8)))  _Float16 h8;
typedef __attribute__((ext_vector_type(16))) _Float16 h16;
typedef __attribute__((ext_vector_type(8)))  float    f8;

__device__ __forceinline__ h16 cat8(h8 lo, h8 hi) {
  return __builtin_shufflevector(lo, hi, 0, 1, 2, 3, 4, 5, 6, 7, 8, 9, 10, 11, 12, 13, 14, 15);
}

// A operand (16xK row-major, ld elems). Lane L holds row M=L&15.
// v16h elems 0-7  <- K = k0 + 8*half .. +7
// v16h elems 8-15 <- K = k0 + 16 + 8*half .. +7   (ISA 16-bit A 16x32 layout)
__device__ __forceinline__ h16 load_a(const _Float16* A, int ld, int k0, int lane) {
  const int row = lane & 15, h = (lane >> 4) & 1;
  const _Float16* p = A + (size_t)row * ld + k0 + 8 * h;
  h8 lo = *(const h8*)p;
  h8 hi = *(const h8*)(p + 16);
  return cat8(lo, hi);
}

// B operand from B^T row-major [N][K]. Lane L holds col N=L&15.
// v16h elems 0-7 <- K = k0 + 8*half .. +7, elems 8-15 <- K = k0 + 16 + 8*half .. +7   (ISA 16-bit B 32x16 layout)
__device__ __forceinline__ h16 load_b(const _Float16* Bt, int ld, int k0, int lane) {
  const int col = lane & 15, h = (lane >> 4) & 1;
  const _Float16* p = Bt + (size_t)col * ld + k0 + 8 * h;
  h8 lo = *(const h8*)p;
  h8 hi = *(const h8*)(p + 16);
  return cat8(lo, hi);
}

__device__ __forceinline__ f8 wmma16(h16 a, h16 b, f8 c) {
  return __builtin_amdgcn_wmma_f32_16x16x32_f16(false, a, false, b, (short)0, c, false, false);
}

// ---------------- conversion kernels ----------------

__global__ __launch_bounds__(256) void cvt_f16_kernel(const float* __restrict__ in,
                                                      _Float16* __restrict__ out, int n) {
  int i = blockIdx.x * 256 + threadIdx.x;
  if (i < n) out[i] = (_Float16)in[i];
}

// out[f][m] = (f16) in[m][f], 1024x1024, LDS-tiled for coalescing
__global__ __launch_bounds__(256) void transpose_f16_kernel(const float* __restrict__ in,
                                                            _Float16* __restrict__ out) {
  __shared__ float t[32][33];
  const int x = threadIdx.x, y = threadIdx.y;
  const int bx = blockIdx.x * 32, by = blockIdx.y * 32;
#pragma unroll
  for (int j = 0; j < 32; j += 8) t[y + j][x] = in[(size_t)(by + y + j) * 1024 + bx + x];
  __syncthreads();
#pragma unroll
  for (int j = 0; j < 32; j += 8)
    out[(size_t)(bx + y + j) * 1024 + by + x] = (_Float16)t[x][y + j];
}

// ---------------- position-bias kernel ----------------
// W[c][n][m] = log(max(pos_w[c,:].emb[:,n,m] + pos_b[c], 1e-6)), c=g=16, e=64
__global__ __launch_bounds__(256) void pos_kernel(const float* __restrict__ emb,   // [64][1024][1024]
                                                  const _Float16* __restrict__ pwh, // [16][64]
                                                  const float* __restrict__ pos_b,  // [16]
                                                  float* __restrict__ W) {          // [16][1024][1024]
  __shared__ __align__(16) _Float16 Bt[8][1024];  // per-wave [m(16)][e(64)]
  const int n = blockIdx.x;
  const int wave = threadIdx.x >> 5, lane = threadIdx.x & 31;
  const int m0 = (blockIdx.y * 8 + wave) * 16;
  _Float16* bt = &Bt[wave][0];
#pragma unroll
  for (int it = 0; it < 8; ++it) {            // stage 64e x 16m, transpose to [m][e] f16
    const int e = it * 8 + (lane >> 2);
    const int mq = (lane & 3) * 4;
    const float4 v = *(const float4*)(emb + (size_t)e * 1048576 + (size_t)n * 1024 + m0 + mq);
    bt[(mq + 0) * 64 + e] = (_Float16)v.x;
    bt[(mq + 1) * 64 + e] = (_Float16)v.y;
    bt[(mq + 2) * 64 + e] = (_Float16)v.z;
    bt[(mq + 3) * 64 + e] = (_Float16)v.w;
  }
  __syncthreads();
  f8 c = {};
#pragma unroll
  for (int k0 = 0; k0 < 64; k0 += 32) {
    h16 a = load_a(pwh, 64, k0, lane);
    h16 b = load_b(bt, 64, k0, lane);
    c = wmma16(a, b, c);
  }
  const int col = lane & 15, rb = (lane >> 4) * 8;
#pragma unroll
  for (int v = 0; v < 8; ++v) {
    const int cr = rb + v;
    const float val = fmaxf(c[v] + pos_b[cr], 1e-6f);  // max(relu(x),1e-6) == max(x,1e-6)
    W[((size_t)cr * 1024 + n) * 1024 + m0 + col] = __logf(val);
  }
}

// ---------------- q/k projection: C = A(1024xF) @ Bt(1024xF)^T + bias, stored [g][n][64] f16
__global__ __launch_bounds__(256) void qk_kernel(const _Float16* __restrict__ A,   // roi f16 [n][f]
                                                 const _Float16* __restrict__ Bt,  // weight f16 [o][f]
                                                 const float* __restrict__ bias,   // [1024]
                                                 _Float16* __restrict__ outg) {    // [16][1024][64]
  const int wave = threadIdx.x >> 5, lane = threadIdx.x & 31;
  const int tile = blockIdx.x * 8 + wave;     // 4096 tiles
  const int n0 = (tile >> 6) * 16, o0 = (tile & 63) * 16;
  f8 c = {};
  for (int k0 = 0; k0 < 1024; k0 += 32) {
    h16 a = load_a(A + (size_t)n0 * 1024, 1024, k0, lane);
    h16 b = load_b(Bt + (size_t)o0 * 1024, 1024, k0, lane);
    c = wmma16(a, b, c);
  }
  const int col = lane & 15, rb = (lane >> 4) * 8;
  const int o = o0 + col, g = o >> 6, d = o & 63;
  const float bi = bias[o];
#pragma unroll
  for (int v = 0; v < 8; ++v) {
    const int n = n0 + rb + v;
    outg[((size_t)g * 1024 + n) * 64 + d] = (_Float16)(c[v] + bi);
  }
}

// ---------------- fused attention: logits -> softmax -> P@V -> per-head out GEMM
__global__ __launch_bounds__(256) void attn_kernel(const float* __restrict__ W,     // [16][1024][1024]
                                                   const _Float16* __restrict__ qh, // [16][1024][64]
                                                   const _Float16* __restrict__ kh, // [16][1024][64]
                                                   const _Float16* __restrict__ vt, // [f][m] f16
                                                   const _Float16* __restrict__ owh,// [o][f] f16
                                                   const float* __restrict__ out_b, // [1024]
                                                   float* __restrict__ out) {       // [1024][1024]
  __shared__ __align__(16) float    S[16 * 1024];  // 64KB logits / exp
  __shared__ __align__(16) _Float16 P[16 * 1024];  // 32KB probs
  _Float16* O = (_Float16*)S;                      // 32KB, reuses dead S region

  const int g = blockIdx.x & 15;
  const int n0 = (blockIdx.x >> 4) * 16;
  const int wave = threadIdx.x >> 5, lane = threadIdx.x & 31;
  const int col = lane & 15, rb = (lane >> 4) * 8;

  const _Float16* qg = qh + ((size_t)g * 1024 + n0) * 64;
  const _Float16* kg = kh + (size_t)g * 1024 * 64;

  // Phase 1: S[n][m] = 0.125 * q.k + logW
  for (int mt = wave; mt < 64; mt += 8) {
    const int m0 = mt * 16;
    f8 c = {};
#pragma unroll
    for (int k0 = 0; k0 < 64; k0 += 32) {
      h16 a = load_a(qg, 64, k0, lane);
      h16 b = load_b(kg + (size_t)m0 * 64, 64, k0, lane);
      c = wmma16(a, b, c);
    }
#pragma unroll
    for (int v = 0; v < 8; ++v) {
      const int n = rb + v;
      const float w = W[((size_t)g * 1024 + n0 + n) * 1024 + m0 + col];
      S[n * 1024 + m0 + col] = c[v] * 0.125f + w;
    }
  }
  __syncthreads();

  // Phase 2: row softmax (2 rows per wave), P = f16 probs
  for (int r = wave; r < 16; r += 8) {
    float mx = -3.0e38f;
    for (int j = lane; j < 1024; j += 32) mx = fmaxf(mx, S[r * 1024 + j]);
#pragma unroll
    for (int off = 16; off; off >>= 1) mx = fmaxf(mx, __shfl_xor(mx, off, 32));
    float sum = 0.f;
    for (int j = lane; j < 1024; j += 32) {
      const float e = __expf(S[r * 1024 + j] - mx);
      S[r * 1024 + j] = e;
      sum += e;
    }
#pragma unroll
    for (int off = 16; off; off >>= 1) sum += __shfl_xor(sum, off, 32);
    const float inv = 1.0f / sum;
    for (int j = lane; j < 1024; j += 32) P[r * 1024 + j] = (_Float16)(S[r * 1024 + j] * inv);
  }
  __syncthreads();

  // Phase 3: O[16][1024] = P @ V ; each wave owns 8 f-tiles (128 cols), A reused across tiles
  f8 acc[8] = {};
  for (int k0 = 0; k0 < 1024; k0 += 32) {
    h16 a = load_a(P, 1024, k0, lane);
#pragma unroll
    for (int t = 0; t < 8; ++t) {
      const int f0 = (wave * 8 + t) * 16;
      h16 b = load_b(vt + (size_t)f0 * 1024, 1024, k0, lane);
      acc[t] = wmma16(a, b, acc[t]);
    }
  }
#pragma unroll
  for (int t = 0; t < 8; ++t) {
    const int f0 = (wave * 8 + t) * 16;
#pragma unroll
    for (int v = 0; v < 8; ++v) O[(rb + v) * 1024 + f0 + col] = (_Float16)acc[t][v];
  }
  __syncthreads();

  // Phase 4: out[n0..+15][g*64 .. +63] = O @ ow[g]^T + out_b (waves 0-3, one 16-col tile each)
  if (wave < 4) {
    const int o0 = wave * 16;
    f8 c = {};
    for (int k0 = 0; k0 < 1024; k0 += 32) {
      h16 a = load_a(O, 1024, k0, lane);
      h16 b = load_b(owh + (size_t)(g * 64 + o0) * 1024, 1024, k0, lane);
      c = wmma16(a, b, c);
    }
    const float bi = out_b[g * 64 + o0 + col];
#pragma unroll
    for (int v = 0; v < 8; ++v)
      out[(size_t)(n0 + rb + v) * 1024 + g * 64 + o0 + col] = c[v] + bi;
  }
}

// ---------------- launcher ----------------
extern "C" void kernel_launch(void* const* d_in, const int* in_sizes, int n_in,
                              void* d_out, int out_size, void* d_ws, size_t ws_size,
                              hipStream_t stream) {
  (void)in_sizes; (void)n_in; (void)out_size; (void)ws_size;
  const float* roi   = (const float*)d_in[0];
  const float* emb   = (const float*)d_in[1];   // [1][64][1024][1024]
  const float* pos_w = (const float*)d_in[3];
  const float* pos_b = (const float*)d_in[4];
  const float* q_w   = (const float*)d_in[5];
  const float* q_b   = (const float*)d_in[6];
  const float* k_w   = (const float*)d_in[7];
  const float* k_b   = (const float*)d_in[8];
  const float* out_w = (const float*)d_in[9];
  const float* out_b = (const float*)d_in[10];

  char* ws = (char*)d_ws;
  float*    W     = (float*)ws;                              // 64 MB  [16][1024][1024]
  _Float16* roi_h = (_Float16*)(ws + (size_t)67108864);      // 2 MB   [n][f]
  _Float16* vt_h  = roi_h + 1048576;                         // 2 MB   [f][m]
  _Float16* qw_h  = vt_h + 1048576;                          // 2 MB   [o][f]
  _Float16* kw_h  = qw_h + 1048576;                          // 2 MB
  _Float16* ow_h  = kw_h + 1048576;                          // 2 MB
  _Float16* q_h   = ow_h + 1048576;                          // 2 MB   [16][1024][64]
  _Float16* k_h   = q_h + 1048576;                           // 2 MB
  _Float16* pw_h  = k_h + 1048576;                           // 2 KB   [16][64]

  const int NF = 1024 * 1024;
  cvt_f16_kernel<<<NF / 256, 256, 0, stream>>>(roi, roi_h, NF);
  cvt_f16_kernel<<<NF / 256, 256, 0, stream>>>(q_w, qw_h, NF);
  cvt_f16_kernel<<<NF / 256, 256, 0, stream>>>(k_w, kw_h, NF);
  cvt_f16_kernel<<<NF / 256, 256, 0, stream>>>(out_w, ow_h, NF);
  cvt_f16_kernel<<<4, 256, 0, stream>>>(pos_w, pw_h, 1024);
  transpose_f16_kernel<<<dim3(32, 32), dim3(32, 8), 0, stream>>>(roi, vt_h);

  pos_kernel<<<dim3(1024, 8), 256, 0, stream>>>(emb, pw_h, pos_b, W);
  qk_kernel<<<512, 256, 0, stream>>>(roi_h, qw_h, q_b, q_h);
  qk_kernel<<<512, 256, 0, stream>>>(roi_h, kw_h, k_b, k_h);
  attn_kernel<<<1024, 256, 0, stream>>>(W, q_h, k_h, vt_h, ow_h, out_b, (float*)d_out);
}
